// BahdanauAttention_28166395527367
// MI455X (gfx1250) — compile-verified, matrix operand order checked by reading, not run on hardware
//
#include <hip/hip_runtime.h>
#include <math.h>

typedef __attribute__((ext_vector_type(16))) __bf16 v16bf;
typedef __attribute__((ext_vector_type(8)))  __bf16 v8bf;
typedef __attribute__((ext_vector_type(4)))  __bf16 v4bf;
typedef __attribute__((ext_vector_type(8)))  float  v8f;

#define BB 32
#define TT 4096
#define DD 512
#define UU 512

// CDNA5 hardware transcendental tanh (v_tanh_f32).
__device__ __forceinline__ float bah_tanh(float x) {
#if __has_builtin(__builtin_amdgcn_tanhf)
    return __builtin_amdgcn_tanhf(x);
#else
    return tanhf(x);
#endif
}

// Butterfly sum over the 16 lanes of each wave half via ds_swizzle
// (group-of-32 XOR mode: offset = {0, xor[4:0], or=0, and=0x1f}).
// The swizzle offset must be an ICE at the call site -> macro with literals.
#if __has_builtin(__builtin_amdgcn_ds_swizzle)
#define BAH_SWZ(x, imm) \
    __builtin_bit_cast(float, \
        __builtin_amdgcn_ds_swizzle(__builtin_bit_cast(int, (x)), (imm)))
__device__ __forceinline__ float bah_xor_sum16(float s) {
    s += BAH_SWZ(s, 0x041F);   // xor 1
    s += BAH_SWZ(s, 0x081F);   // xor 2
    s += BAH_SWZ(s, 0x101F);   // xor 4
    s += BAH_SWZ(s, 0x201F);   // xor 8
    return s;
}
#else
__device__ __forceinline__ float bah_xor_sum16(float s) {
    for (int m = 1; m < 16; m <<= 1) s += __shfl_xor(s, m, 32);
    return s;
}
#endif

// ---------------------------------------------------------------------------
// 1) add[b,u] = (query @ W1)[b,u] + b1[u] + b2[u]
// ---------------------------------------------------------------------------
__global__ void bah_qproj(const float* __restrict__ query,
                          const float* __restrict__ W1,
                          const float* __restrict__ b1,
                          const float* __restrict__ b2,
                          float* __restrict__ addout) {
    __shared__ float q[DD];
    const int b = blockIdx.x;
    const int u = threadIdx.x;          // 512 threads
    q[u] = query[b * DD + u];
    __syncthreads();
    float acc = 0.f;
#pragma unroll 8
    for (int d = 0; d < DD; ++d)
        acc = fmaf(q[d], W1[d * UU + u], acc);
    addout[b * UU + u] = acc + b1[u] + b2[u];
}

// ---------------------------------------------------------------------------
// 2) Pack W2 (f32 row-major [D,U]) into fragment-major bf16 for the WMMA
//    B operand: packed[g], g = ((ct*16 + kt)*32 + lane)*16 + e, with the ISA
//    16-bit B layout:  K = kt*32 + (e&7) + 8*(lane>>4) + 16*(e>>3),
//                      N = ct*16 + (lane&15)
// ---------------------------------------------------------------------------
__global__ void bah_packW2(const float* __restrict__ W2,
                           __bf16* __restrict__ w2p) {
    const int g    = blockIdx.x * 256 + threadIdx.x;   // 512*512 entries
    const int e    = g & 15;
    const int lane = (g >> 4) & 31;
    const int kt   = (g >> 9) & 15;
    const int ct   = g >> 13;
    const int d = kt * 32 + (e & 7) + 8 * (lane >> 4) + 16 * (e >> 3);
    const int u = ct * 16 + (lane & 15);
    w2p[g] = (__bf16)W2[d * UU + u];
}

// ---------------------------------------------------------------------------
// Fused epilogue for two 16x16 v_proj tiles covering the SAME output rows
// (ctA, ctB): tanh(acc+add)*V for both, combine, one butterfly reduction per
// row value, LDS-atomic accumulate into the per-t partial scores.
// ---------------------------------------------------------------------------
__device__ __forceinline__ void bah_epilogue2(const v8f& accA, int ctA,
                                              const v8f& accB, int ctB,
                                              int r, int h, int rbase,
                                              const float* __restrict__ addv_b,
                                              const float* __restrict__ Vvec,
                                              float* __restrict__ lds_s) {
    const int uA = ctA * 16 + r, uB = ctB * 16 + r;
    const float avA = addv_b[uA], vuA = Vvec[uA];
    const float avB = addv_b[uB], vuB = Vvec[uB];
#pragma unroll
    for (int j = 0; j < 8; ++j) {
        float s = bah_tanh(accA[j] + avA) * vuA
                + bah_tanh(accB[j] + avB) * vuB;
        s = bah_xor_sum16(s);
        if (r == 0) atomicAdd(&lds_s[rbase + 8 * h + j], s);
    }
}

// ---------------------------------------------------------------------------
// 3) Main WMMA kernel: per block = one batch x 32 time rows (M=32 macro-tile).
//    Register block = 2 row-frags x 2 col-tiles -> 4 WMMAs per kt step from
//    2 A + 2 B fragment loads; all B addresses are immediate offsets from one
//    hoisted per-lane base pointer.
// ---------------------------------------------------------------------------
__global__ void __launch_bounds__(256)
bah_scores(const float* __restrict__ values,
           const __bf16* __restrict__ w2p,
           const float* __restrict__ addv,    // [B,U]
           const float* __restrict__ Vvec,    // [U]
           const float* __restrict__ bv,      // [1]
           float* __restrict__ scores) {      // [B,T]
    __shared__ __bf16 ldsA[32 * DD];          // 32 KB bf16 tile (32 t-rows)
    __shared__ float  lds_s[32];

    const int blk = blockIdx.x;               // B * T/32 = 4096 blocks
    const int b   = blk >> 7;                 // 128 tiles per batch
    const int t0  = (blk & 127) * 32;
    const int tid = threadIdx.x;

    // Stage the contiguous 32x512 f32 tile as bf16 into LDS (coalesced b128).
    const float4* src = (const float4*)(values + ((size_t)b * TT + t0) * DD);
#pragma unroll
    for (int k = 0; k < 16; ++k) {
        const int i4 = tid + k * 256;         // 0..4095 float4s
        float4 v = src[i4];
        v4bf p;
        p[0] = (__bf16)v.x; p[1] = (__bf16)v.y;
        p[2] = (__bf16)v.z; p[3] = (__bf16)v.w;
        *(v4bf*)&ldsA[i4 * 4] = p;
    }
    if (tid < 32) lds_s[tid] = 0.f;
    __syncthreads();

    const int w    = tid >> 5;
    const int lane = tid & 31;
    const int h    = lane >> 4;               // K-half select (A layout)
    const int r    = lane & 15;               // A row / B column index
    const float* addv_b = addv + b * UU;

    // Hoisted per-lane B base: element offset = ct*8192 + kt*512 + lane*16.
    const __bf16* wpl = w2p + (size_t)(w * 8192 + lane * 16);
    const int abase0 = r * DD + 8 * h;

#pragma unroll
    for (int p = 0; p < 2; ++p) {             // wave w owns ct {w,w+8,w+16,w+24}
        v8f acc00 = {}, acc01 = {}, acc10 = {}, acc11 = {};
#pragma unroll 2
        for (int kt = 0; kt < 16; ++kt) {
            const int ab = abase0 + kt * 32;
            // Two A row-fragments (rows 0-15 and 16-31), immediate DS offsets
            v8bf lo0 = *(const v8bf*)&ldsA[ab];
            v8bf hi0 = *(const v8bf*)&ldsA[ab + 16];
            v8bf lo1 = *(const v8bf*)&ldsA[ab + 16 * DD];
            v8bf hi1 = *(const v8bf*)&ldsA[ab + 16 * DD + 16];
            v16bf a0 = __builtin_shufflevector(lo0, hi0,
                0, 1, 2, 3, 4, 5, 6, 7, 8, 9, 10, 11, 12, 13, 14, 15);
            v16bf a1 = __builtin_shufflevector(lo1, hi1,
                0, 1, 2, 3, 4, 5, 6, 7, 8, 9, 10, 11, 12, 13, 14, 15);
            // Two B col-fragments, immediate global offsets from wpl
            v16bf b0 = *(const v16bf*)(wpl + p * 131072 + kt * 512);
            v16bf b1 = *(const v16bf*)(wpl + p * 131072 + kt * 512 + 65536);
            acc00 = __builtin_amdgcn_wmma_f32_16x16x32_bf16(
                false, a0, false, b0, (short)0, acc00, false, false);
            acc10 = __builtin_amdgcn_wmma_f32_16x16x32_bf16(
                false, a1, false, b0, (short)0, acc10, false, false);
            acc01 = __builtin_amdgcn_wmma_f32_16x16x32_bf16(
                false, a0, false, b1, (short)0, acc01, false, false);
            acc11 = __builtin_amdgcn_wmma_f32_16x16x32_bf16(
                false, a1, false, b1, (short)0, acc11, false, false);
        }
        const int ct0 = w + p * 16;
        const int ct1 = ct0 + 8;
        bah_epilogue2(acc00, ct0, acc01, ct1, r, h,  0, addv_b, Vvec, lds_s);
        bah_epilogue2(acc10, ct0, acc11, ct1, r, h, 16, addv_b, Vvec, lds_s);
    }
    __syncthreads();
    if (tid < 32)
        scores[(size_t)b * TT + t0 + tid] = lds_s[tid] + bv[0];
}

// ---------------------------------------------------------------------------
// 4) Per-batch softmax over T: weights[b,t] = exp(s - max) / sum
// ---------------------------------------------------------------------------
__global__ void bah_softmax(const float* __restrict__ scores,
                            float* __restrict__ weights) {
    __shared__ float red[256];
    const int b = blockIdx.x, tid = threadIdx.x;
    const float* s = scores + (size_t)b * TT;

    float lm = -3.4e38f;
    for (int t = tid; t < TT; t += 256) lm = fmaxf(lm, s[t]);
    red[tid] = lm; __syncthreads();
    for (int o = 128; o > 0; o >>= 1) {
        if (tid < o) red[tid] = fmaxf(red[tid], red[tid + o]);
        __syncthreads();
    }
    const float m = red[0];
    __syncthreads();

    float ls = 0.f;
    for (int t = tid; t < TT; t += 256) ls += __expf(s[t] - m);
    red[tid] = ls; __syncthreads();
    for (int o = 128; o > 0; o >>= 1) {
        if (tid < o) red[tid] += red[tid + o];
        __syncthreads();
    }
    const float inv = 1.0f / red[0];
    __syncthreads();

    for (int t = tid; t < TT; t += 256)
        weights[(size_t)b * TT + t] = __expf(s[t] - m) * inv;
}

// ---------------------------------------------------------------------------
// 5) context partials over 16 t-chunks (deterministic, no FP atomics)
// ---------------------------------------------------------------------------
__global__ void bah_ctx_partial(const float* __restrict__ values,
                                const float* __restrict__ weights,
                                float* __restrict__ partial) {  // [B,16,D]
    const int blk = blockIdx.x;         // 32 * 2 * 16 = 1024
    const int b  = blk >> 5;
    const int dc = (blk >> 4) & 1;
    const int tc = blk & 15;
    const int d  = dc * 256 + threadIdx.x;

    const float* wb = weights + (size_t)b * TT + tc * 256;
    const float* vb = values + ((size_t)b * TT + tc * 256) * DD + d;
    float acc = 0.f;
#pragma unroll 4
    for (int t = 0; t < 256; ++t)
        acc = fmaf(wb[t], vb[(size_t)t * DD], acc);
    partial[((size_t)(b * 16 + tc)) * DD + d] = acc;
}

__global__ void bah_ctx_reduce(const float* __restrict__ partial,
                               float* __restrict__ out) {       // [B,D]
    const int g = blockIdx.x * 256 + threadIdx.x;  // 16384
    const int b = g >> 9, d = g & 511;
    float acc = 0.f;
#pragma unroll
    for (int tc = 0; tc < 16; ++tc)
        acc += partial[((size_t)(b * 16 + tc)) * DD + d];
    out[g] = acc;
}

// ---------------------------------------------------------------------------
extern "C" void kernel_launch(void* const* d_in, const int* in_sizes, int n_in,
                              void* d_out, int out_size, void* d_ws, size_t ws_size,
                              hipStream_t stream) {
    (void)in_sizes; (void)n_in; (void)out_size; (void)ws_size;
    const float* query = (const float*)d_in[0];
    const float* values= (const float*)d_in[1];
    const float* W1    = (const float*)d_in[2];
    const float* b1    = (const float*)d_in[3];
    const float* W2    = (const float*)d_in[4];
    const float* b2    = (const float*)d_in[5];
    const float* Vv    = (const float*)d_in[6];
    const float* bv    = (const float*)d_in[7];
    float* out = (float*)d_out;

    char* ws = (char*)d_ws;
    float*  addv    = (float*)(ws);                     //  64 KB  [B,U]
    __bf16* w2p     = (__bf16*)(ws + 65536);            // 512 KB  packed W2
    float*  scores  = (float*)(ws + 589824);            // 512 KB  [B,T]
    float*  weights = (float*)(ws + 1114112);           // 512 KB  [B,T]
    float*  partial = (float*)(ws + 1638400);           //   1 MB  [B,16,D]

    bah_qproj      <<<BB,            512, 0, stream>>>(query, W1, b1, b2, addv);
    bah_packW2     <<<(DD*UU)/256,   256, 0, stream>>>(W2, w2p);
    bah_scores     <<<BB*(TT/32),    256, 0, stream>>>(values, w2p, addv, Vv, bv, scores);
    bah_softmax    <<<BB,            256, 0, stream>>>(scores, weights);
    bah_ctx_partial<<<BB*2*16,       256, 0, stream>>>(values, weights, partial);
    bah_ctx_reduce <<<(BB*DD)/256,   256, 0, stream>>>(partial, out);
}
